// Self_Attention_51565377356498
// MI455X (gfx1250) — matrix operand order repaired and checked by reading, not compiled, not run on hardware
//
#include <hip/hip_runtime.h>
#include <hip/hip_bf16.h>
#include <stdint.h>

// ---------------------------------------------------------------------------
// CDNA5 WMMA fragment helpers (v_wmma_f32_16x16x32_f16, wave32)
// ---------------------------------------------------------------------------
typedef __attribute__((ext_vector_type(16))) _Float16 v16h;
typedef __attribute__((ext_vector_type(8)))  float    v8f;

union Frag16 { v16h h; uint32_t u[8]; };

// A-matrix 16x32 f16 layout (ISA 7.12.2): lane holds row m=lane&15;
// VGPR v holds K pair starting at:
__device__ __forceinline__ int aK(int v, int lane) {
    return ((v & 4) << 2) + ((lane >> 4) << 3) + ((v & 3) << 1);
}
// B-matrix 32x16 f16 layout: lane holds column n=lane&15;
// same K order as A: VGPR v holds K pair starting at aK(v, lane).
__device__ __forceinline__ int bK(int v, int lane) {
    return ((v & 4) << 2) + ((lane >> 4) << 3) + ((v & 3) << 1);
}

__device__ __forceinline__ v8f wmma_f16(const Frag16& a, const Frag16& b, v8f c) {
    return __builtin_amdgcn_wmma_f32_16x16x32_f16(
        /*neg_a=*/false, a.h, /*neg_b=*/false, b.h,
        /*c_mod=*/(short)0, c, /*reuse_a=*/false, /*reuse_b=*/false);
}

// ---------------------------------------------------------------------------
// Async global->LDS staging (gfx1250 GLOBAL_LOAD_ASYNC_TO_LDS_B128, ASYNCcnt)
// Builtin signature (from compiler diagnostic):
//   (int4 addrspace(1)*, int4 addrspace(3)*, imm offset, imm cpol)
// Falls back to VGPR round-trip if this toolchain lacks the builtin.
// ---------------------------------------------------------------------------
typedef int vint4 __attribute__((vector_size(16)));
typedef __attribute__((address_space(1))) vint4 g_vint4;
typedef __attribute__((address_space(3))) vint4 l_vint4;

#if __has_builtin(__builtin_amdgcn_global_load_async_to_lds_b128)
#define HAVE_ASYNC_LDS 1
#endif

__device__ __forceinline__ void async_b128(const _Float16* g, _Float16* l) {
#ifdef HAVE_ASYNC_LDS
    __builtin_amdgcn_global_load_async_to_lds_b128(
        (g_vint4*)(void*)g, (l_vint4*)(void*)l, /*offset=*/0, /*cpol=*/0);
#else
    *(uint4*)l = *(const uint4*)g;
#endif
}

__device__ __forceinline__ void wait_async0() {
#ifdef HAVE_ASYNC_LDS
#if __has_builtin(__builtin_amdgcn_s_wait_asynccnt)
    __builtin_amdgcn_s_wait_asynccnt(0);
#else
    asm volatile("s_wait_asynccnt 0x0" ::: "memory");
#endif
#endif
}

// ---------------------------------------------------------------------------
// fp32 -> f16 conversion
// ---------------------------------------------------------------------------
__global__ void cvt_f32_to_f16(const float* __restrict__ in,
                               _Float16* __restrict__ out, int n) {
    int i = blockIdx.x * blockDim.x + threadIdx.x;
    if (i < n) out[i] = (_Float16)in[i];
}

// ---------------------------------------------------------------------------
// Tiled WMMA GEMM: C[4096x1024] = A[4096x1024] * B[1024x1024], f16 in, f32 acc
// Block tile 128x128, 8 waves (4 along M x 2 along N), each wave 32x64.
// MODE 0: QKV projection, f16 output scattered to [B,H,L,64], blockIdx.z picks W.
// MODE 1: output projection, f32 output row-major to d_out.
// ---------------------------------------------------------------------------
template<int MODE>
__global__ __launch_bounds__(256) void gemm128(const _Float16* __restrict__ A,
                                               const _Float16* __restrict__ Wbase,
                                               _Float16* __restrict__ dq,
                                               _Float16* __restrict__ dk,
                                               _Float16* __restrict__ dv,
                                               float* __restrict__ outf) {
    constexpr int Kdim = 1024, N = 1024;
    __shared__ _Float16 sA[128 * 32];      // [m][k]
    __shared__ _Float16 sB[128 * 36];      // transposed [n][k], padded stride 36

    const _Float16* B;
    _Float16* dsth = nullptr;
    if (MODE == 0) {
        B = Wbase + (size_t)blockIdx.z * (1024u * 1024u);
        dsth = (blockIdx.z == 0) ? dq : ((blockIdx.z == 1) ? dk : dv);
    } else {
        B = Wbase;
    }

    const int tid = threadIdx.x;
    const int lane = tid & 31, wave = tid >> 5;
    const int waveM = wave & 3, waveN = wave >> 2;
    const int row0 = blockIdx.y * 128;
    const int col0 = blockIdx.x * 128;

    v8f zero = {};
    v8f acc[2][4];
#pragma unroll
    for (int mt = 0; mt < 2; ++mt)
#pragma unroll
        for (int nt = 0; nt < 4; ++nt) acc[mt][nt] = zero;

    for (int kb = 0; kb < Kdim / 32; ++kb) {
        {   // stage A tile 128x32 (row-major, k contiguous) via async->LDS
            int r = tid >> 1;
            int koff = (tid & 1) * 16;
            const _Float16* src = A + (size_t)(row0 + r) * Kdim + kb * 32 + koff;
            _Float16* dst = sA + r * 32 + koff;
            async_b128(src, dst);
            async_b128(src + 8, dst + 8);
        }
        {   // stage B tile 32x128, transposed into [n][k] (manual: async can't transpose)
            int k = tid >> 3;
            int n0 = (tid & 7) * 16;
            const _Float16* src = B + (size_t)(kb * 32 + k) * N + col0 + n0;
#pragma unroll
            for (int i = 0; i < 16; ++i) sB[(n0 + i) * 36 + k] = src[i];
        }
        wait_async0();
        __syncthreads();

        Frag16 af[2], bf[4];
#pragma unroll
        for (int mt = 0; mt < 2; ++mt) {
            int r = waveM * 32 + mt * 16 + (lane & 15);
#pragma unroll
            for (int v = 0; v < 8; ++v)
                af[mt].u[v] = *(const uint32_t*)(sA + r * 32 + aK(v, lane));
        }
#pragma unroll
        for (int nt = 0; nt < 4; ++nt) {
            int n = waveN * 64 + nt * 16 + (lane & 15);
#pragma unroll
            for (int v = 0; v < 8; ++v)
                bf[nt].u[v] = *(const uint32_t*)(sB + n * 36 + bK(v, lane));
        }
#pragma unroll
        for (int mt = 0; mt < 2; ++mt)
#pragma unroll
            for (int nt = 0; nt < 4; ++nt)
                acc[mt][nt] = wmma_f16(af[mt], bf[nt], acc[mt][nt]);
        __syncthreads();
    }

    // Epilogue: C layout m = r + 8*(lane>>4), n = lane&15
    const int g = lane >> 4, nl = lane & 15;
#pragma unroll
    for (int mt = 0; mt < 2; ++mt)
#pragma unroll
        for (int nt = 0; nt < 4; ++nt)
#pragma unroll
            for (int r = 0; r < 8; ++r) {
                int row = row0 + waveM * 32 + mt * 16 + r + g * 8;
                int col = col0 + waveN * 64 + nt * 16 + nl;
                float val = acc[mt][nt][r];
                if (MODE == 0) {
                    int b = row >> 11, l = row & 2047;   // M = B*L, L = 2048
                    int hh = col >> 6, d = col & 63;     // N = H*64
                    dsth[(((size_t)(b * 16 + hh)) * 2048 + l) * 64 + d] = (_Float16)val;
                } else {
                    outf[(size_t)row * 1024 + col] = val;
                }
            }
}

// ---------------------------------------------------------------------------
// Flash attention: block = (b, h, 64 q-rows); 4 waves x 16-row Q strips.
// Streams K/V in 32-row tiles through LDS; online softmax; PE added fp32.
// ---------------------------------------------------------------------------
__global__ __launch_bounds__(128) void attn_kernel(const _Float16* __restrict__ qh,
                                                   const _Float16* __restrict__ kh,
                                                   const _Float16* __restrict__ vh,
                                                   const float* __restrict__ pe,
                                                   _Float16* __restrict__ aout) {
    constexpr int L = 2048, Hd = 64;
    __shared__ _Float16 sK[32 * 64];       // K tile [kr][d], natural
    __shared__ _Float16 sVT[64 * 34];      // V tile transposed [d][kr], padded
    __shared__ _Float16 sP[4][16 * 32];    // per-wave P staging (C->A transpose)

    const int tid = threadIdx.x, lane = tid & 31, wave = tid >> 5;
    const int qb = blockIdx.x & 31;
    const int h  = (blockIdx.x >> 5) & 15;
    const int b  = blockIdx.x >> 9;
    const int headBase = (b * 16 + h) * L;
    const int q0 = qb * 64 + wave * 16;
    const int g = lane >> 4, nl = lane & 15;

    // Q fragments (A layout, K-dim = head dim, split 0..31 / 32..63)
    Frag16 qa0, qa1;
    {
        const _Float16* qrow = qh + ((size_t)headBase + q0 + nl) * Hd;
#pragma unroll
        for (int v = 0; v < 8; ++v) {
            int d0 = aK(v, lane);
            qa0.u[v] = *(const uint32_t*)(qrow + d0);
            qa1.u[v] = *(const uint32_t*)(qrow + 32 + d0);
        }
    }

    float mrow[8], lrow[8];
    v8f zero = {};
    v8f o[4];
#pragma unroll
    for (int r = 0; r < 8; ++r) { mrow[r] = -3.0e38f; lrow[r] = 0.0f; }
#pragma unroll
    for (int nt = 0; nt < 4; ++nt) o[nt] = zero;

    const float* peQ = pe + ((size_t)h * L + q0) * L;

    for (int j = 0; j < 64; ++j) {        // 64 tiles of 32 K/V rows
        {   // cooperative stage: K natural via async->LDS, V transposed manually
            int kr = tid >> 2, d0 = (tid & 3) * 16;
            const _Float16* ksrc = kh + ((size_t)headBase + j * 32 + kr) * Hd + d0;
            _Float16* kdst = sK + kr * 64 + d0;
            async_b128(ksrc, kdst);
            async_b128(ksrc + 8, kdst + 8);
            const _Float16* vsrc = vh + ((size_t)headBase + j * 32 + kr) * Hd + d0;
#pragma unroll
            for (int i = 0; i < 16; ++i) sVT[(d0 + i) * 34 + kr] = vsrc[i];
        }
        wait_async0();
        __syncthreads();

        // S = Q @ K^T  (two 16x16 column tiles, K-dim 64 = 2 WMMAs each)
        v8f s[2];
#pragma unroll
        for (int ct = 0; ct < 2; ++ct) {
            Frag16 kb0, kb1;
            const _Float16* krow = sK + (ct * 16 + nl) * 64;
#pragma unroll
            for (int v = 0; v < 8; ++v) {
                int d0 = bK(v, lane);
                kb0.u[v] = *(const uint32_t*)(krow + d0);
                kb1.u[v] = *(const uint32_t*)(krow + 32 + d0);
            }
            v8f t = wmma_f16(qa0, kb0, zero);
            s[ct] = wmma_f16(qa1, kb1, t);
        }

        // scale + position embedding + online softmax (rows live per-VGPR)
#pragma unroll
        for (int r = 0; r < 8; ++r) {
            int m = r + g * 8;
            const float* per = peQ + (size_t)m * L + j * 32;
            float s0 = s[0][r] * 0.125f + per[nl];
            float s1 = s[1][r] * 0.125f + per[16 + nl];
            float rm = fmaxf(s0, s1);
#pragma unroll
            for (int off = 1; off < 16; off <<= 1)
                rm = fmaxf(rm, __shfl_xor(rm, off, 32));
            float mnew = fmaxf(mrow[r], rm);
            float fac = __expf(mrow[r] - mnew);
            float p0 = __expf(s0 - mnew);
            float p1 = __expf(s1 - mnew);
            float rs = p0 + p1;
#pragma unroll
            for (int off = 1; off < 16; off <<= 1)
                rs += __shfl_xor(rs, off, 32);
            lrow[r] = lrow[r] * fac + rs;
            mrow[r] = mnew;
#pragma unroll
            for (int nt = 0; nt < 4; ++nt) o[nt][r] *= fac;
            sP[wave][m * 32 + nl] = (_Float16)p0;        // C-layout -> LDS
            sP[wave][m * 32 + 16 + nl] = (_Float16)p1;
        }

        // O += P @ V   (P re-read in A layout; V^T gives B fragments)
        Frag16 pa;
        {
            const _Float16* prow = &sP[wave][nl * 32];
#pragma unroll
            for (int v = 0; v < 8; ++v)
                pa.u[v] = *(const uint32_t*)(prow + aK(v, lane));
        }
#pragma unroll
        for (int nt = 0; nt < 4; ++nt) {
            Frag16 vb;
            const _Float16* vcol = sVT + (nt * 16 + nl) * 34;
#pragma unroll
            for (int v = 0; v < 8; ++v)
                vb.u[v] = *(const uint32_t*)(vcol + bK(v, lane));
            o[nt] = wmma_f16(pa, vb, o[nt]);
        }
        __syncthreads();
    }

    // normalize + write f16 to [B, L, H*64] for the output projection
#pragma unroll
    for (int r = 0; r < 8; ++r) {
        int m = r + g * 8;
        float inv = 1.0f / lrow[r];
        int l = q0 + m;
#pragma unroll
        for (int nt = 0; nt < 4; ++nt) {
            int col = h * 64 + nt * 16 + nl;
            aout[((size_t)b * L + l) * 1024 + col] = (_Float16)(o[nt][r] * inv);
        }
    }
}

// ---------------------------------------------------------------------------
// Launch
// ---------------------------------------------------------------------------
extern "C" void kernel_launch(void* const* d_in, const int* in_sizes, int n_in,
                              void* d_out, int out_size, void* d_ws, size_t ws_size,
                              hipStream_t stream) {
    (void)in_sizes; (void)n_in; (void)out_size; (void)ws_size;
    const float* x  = (const float*)d_in[0];
    const float* pe = (const float*)d_in[1];
    const float* Wq = (const float*)d_in[2];
    const float* Wk = (const float*)d_in[3];
    const float* Wv = (const float*)d_in[4];
    const float* Wo = (const float*)d_in[5];
    float* out = (float*)d_out;

    char* ws = (char*)d_ws;
    // ws layout (48 MiB total):
    _Float16* xh = (_Float16*)(ws);                    //  8 MiB: x  [4096,1024] f16
    _Float16* wh = (_Float16*)(ws + (8u  << 20));      //  8 MiB: Wq|Wk|Wv|Wo f16
    _Float16* qh = (_Float16*)(ws + (16u << 20));      //  8 MiB: Q [B,H,L,64] f16
    _Float16* kh = (_Float16*)(ws + (24u << 20));      //  8 MiB: K
    _Float16* vh = (_Float16*)(ws + (32u << 20));      //  8 MiB: V
    _Float16* ah = (_Float16*)(ws + (40u << 20));      //  8 MiB: attn out [B,L,1024] f16

    cvt_f32_to_f16<<<4194304 / 256, 256, 0, stream>>>(x,  xh, 4194304);
    cvt_f32_to_f16<<<1048576 / 256, 256, 0, stream>>>(Wq, wh,               1048576);
    cvt_f32_to_f16<<<1048576 / 256, 256, 0, stream>>>(Wk, wh + 1048576,     1048576);
    cvt_f32_to_f16<<<1048576 / 256, 256, 0, stream>>>(Wv, wh + 2 * 1048576, 1048576);
    cvt_f32_to_f16<<<1048576 / 256, 256, 0, stream>>>(Wo, wh + 3 * 1048576, 1048576);

    // QKV projections: grid.z selects Wq/Wk/Wv, scatter to [B,H,L,64]
    gemm128<0><<<dim3(8, 32, 3), 256, 0, stream>>>(xh, wh, qh, kh, vh, nullptr);

    // Flash attention: 2 (B) * 16 (H) * 32 (q-blocks of 64) = 1024 blocks
    attn_kernel<<<1024, 128, 0, stream>>>(qh, kh, vh, pe, ah);

    // Output projection -> fp32 d_out
    gemm128<1><<<dim3(8, 32, 1), 256, 0, stream>>>(ah, wh + 3 * 1048576,
                                                   nullptr, nullptr, nullptr, out);
}
